// MinibatchDiscrimination_55662776156554
// MI455X (gfx1250) — matrix operand order repaired and checked by reading, not compiled, not run on hardware
//
#include <hip/hip_runtime.h>
#include <hip/hip_bf16.h>

// Problem constants
#define Bsz 512
#define Fsz 2048
#define Ksz 64
#define Dsz 16
#define KD  (Ksz * Dsz)          // 1024
#define OUTC (Fsz + Ksz)         // 2112

typedef __attribute__((ext_vector_type(16))) __bf16 v16bf;
typedef __attribute__((ext_vector_type(8)))  float  v8f;
typedef __attribute__((ext_vector_type(4)))  unsigned int v4u;
typedef __attribute__((ext_vector_type(4)))  int v4i;

union Frag16 { v16bf h; v4u u[2]; };

// CDNA5 async global->LDS path (guarded: falls back to sync staging if absent)
#if defined(__has_builtin)
#if __has_builtin(__builtin_amdgcn_global_load_async_to_lds_b128) && \
    __has_builtin(__builtin_amdgcn_s_wait_asynccnt)
#define USE_ASYNC_LDS 1
#endif
#endif

#ifdef USE_ASYNC_LDS
// Signature (from hipcc diagnostic): param0 = AS1 pointer to 128-bit int vector,
// param1 = LDS pointer to same, then imm offset, imm cpol.
typedef __attribute__((address_space(1))) v4i* gptr_b128;
typedef __attribute__((address_space(3))) v4i* lptr_b128;
#endif

__device__ __forceinline__ unsigned short f2bf(float f) {
    unsigned int u = __float_as_uint(f);
    unsigned int r = 0x7FFFu + ((u >> 16) & 1u);   // round-to-nearest-even
    return (unsigned short)((u + r) >> 16);
}

// Kernel 1: x (f32) -> xb (bf16) in ws, and copy x into out[:, 0:2048]
__global__ __launch_bounds__(256) void prep_kernel(const float* __restrict__ x,
                                                   unsigned short* __restrict__ xb,
                                                   float* __restrict__ out) {
    int i = (blockIdx.x * 256 + threadIdx.x) * 4;   // over 512*2048 elems
    float4 v = *(const float4*)(x + i);
    ushort4 h;
    h.x = f2bf(v.x); h.y = f2bf(v.y); h.z = f2bf(v.z); h.w = f2bf(v.w);
    *(ushort4*)(xb + i) = h;
    int b = i / Fsz;
    int f = i % Fsz;
    *(float4*)(out + b * OUTC + f) = v;
}

// Kernel 2: W [F=2048][KD=1024] f32 -> wT [KD=1024][F=2048] bf16 (tiled transpose)
__global__ __launch_bounds__(256) void convert_transpose_kernel(const float* __restrict__ W,
                                                                unsigned short* __restrict__ wT) {
    __shared__ unsigned short tile[32][33];
    int n0 = blockIdx.x * 32;   // kd tile
    int f0 = blockIdx.y * 32;   // f tile
    int tx = threadIdx.x;       // 0..31
    int ty = threadIdx.y;       // 0..7
#pragma unroll
    for (int i = 0; i < 4; ++i) {
        int f = f0 + ty + i * 8;
        tile[ty + i * 8][tx] = f2bf(W[f * KD + n0 + tx]);
    }
    __syncthreads();
#pragma unroll
    for (int i = 0; i < 4; ++i) {
        int n = n0 + ty + i * 8;
        wT[n * Fsz + f0 + tx] = tile[tx][ty + i * 8];
    }
}

// Kernel 3: act[512][1024] = xb[512][2048] @ W via v_wmma_f32_16x16x32_bf16.
// Block = 256 threads = 8 waves; block tile 128(M) x 64(N).
// B (wT rows n0..n0+63, K-chunk of 32) staged in double-buffered LDS, shared by
// all 8 waves; A fragments double-buffered in registers, loaded direct from global.
#define BROW 40   // padded LDS row stride in ushorts (80B) to break bank conflicts
__global__ __launch_bounds__(256) void gemm_wmma_kernel(const unsigned short* __restrict__ xb,
                                                        const unsigned short* __restrict__ wT,
                                                        float* __restrict__ act) {
    __shared__ unsigned short Bsh[2][64 * BROW];   // 2 x 5120B

    const int tid  = threadIdx.x;
    const int wave = tid >> 5;
    const int lane = tid & 31;
    const int m0 = blockIdx.x * 128 + wave * 16;
    const int n0 = blockIdx.y * 64;

    const int lhalf = lane >> 4;        // 0 or 1
    const int l16   = lane & 15;

    // staging assignment: 256 threads cover 64 rows x 4 chunks of 16B
    const int srow   = tid >> 2;        // 0..63
    const int schunk = tid & 3;         // 0..3
    const unsigned short* sgsrc = wT + (n0 + srow) * Fsz + schunk * 8;
    unsigned short* sldst0 = &Bsh[0][srow * BROW + schunk * 8];
    unsigned short* sldst1 = &Bsh[1][srow * BROW + schunk * 8];

    // A fragment source: row m0+l16; lane-half selects K sub-chunks {0..7,16..23} vs {8..15,24..31}
    const unsigned short* aptr = xb + (m0 + l16) * Fsz + lhalf * 8;

    v8f zero = {0.f,0.f,0.f,0.f,0.f,0.f,0.f,0.f};
    v8f acc0 = zero, acc1 = zero, acc2 = zero, acc3 = zero;

#ifdef USE_ASYNC_LDS
#define STAGE_B(dst, kbase)                                                          \
    __builtin_amdgcn_global_load_async_to_lds_b128(                                  \
        (gptr_b128)(void*)(const_cast<unsigned short*>(sgsrc + (kbase))),            \
        (lptr_b128)(void*)(dst), 0, 0)
#define STAGE_WAIT() __builtin_amdgcn_s_wait_asynccnt(0)
#else
#define STAGE_B(dst, kbase)                                                          \
    do { *(v4u*)(dst) = *(const v4u*)(sgsrc + (kbase)); } while (0)
#define STAGE_WAIT() do {} while (0)
#endif

    // prologue: stage k-step 0 into buffer 0, load A fragment 0
    Frag16 a_cur, a_next;
    STAGE_B(sldst0, 0);
    a_cur.u[0] = *(const v4u*)(aptr);
    a_cur.u[1] = *(const v4u*)(aptr + 16);
    STAGE_WAIT();
    __syncthreads();

    for (int ks = 0; ks < Fsz / 32; ++ks) {
        const int k0  = ks * 32;
        const int cur = ks & 1;
        const bool more = (ks + 1) < (Fsz / 32);

        // prefetch next k-step while computing this one
        if (more) {
            STAGE_B(cur ? sldst0 : sldst1, k0 + 32);
            a_next.u[0] = *(const v4u*)(aptr + k0 + 32);
            a_next.u[1] = *(const v4u*)(aptr + k0 + 48);
        }

        // load all four B fragments from LDS, then issue 4 back-to-back WMMAs
        Frag16 b0, b1, b2, b3;
        {
            const unsigned short* base = &Bsh[cur][lhalf * 8];
            const v4u* p0 = (const v4u*)(base + (0 * 16 + l16) * BROW);
            const v4u* p1 = (const v4u*)(base + (1 * 16 + l16) * BROW);
            const v4u* p2 = (const v4u*)(base + (2 * 16 + l16) * BROW);
            const v4u* p3 = (const v4u*)(base + (3 * 16 + l16) * BROW);
            b0.u[0] = p0[0]; b0.u[1] = p0[2];
            b1.u[0] = p1[0]; b1.u[1] = p1[2];
            b2.u[0] = p2[0]; b2.u[1] = p2[2];
            b3.u[0] = p3[0]; b3.u[1] = p3[2];
        }
        acc0 = __builtin_amdgcn_wmma_f32_16x16x32_bf16(false, a_cur.h, false, b0.h,
                                                       (short)0, acc0, false, false);
        acc1 = __builtin_amdgcn_wmma_f32_16x16x32_bf16(false, a_cur.h, false, b1.h,
                                                       (short)0, acc1, false, false);
        acc2 = __builtin_amdgcn_wmma_f32_16x16x32_bf16(false, a_cur.h, false, b2.h,
                                                       (short)0, acc2, false, false);
        acc3 = __builtin_amdgcn_wmma_f32_16x16x32_bf16(false, a_cur.h, false, b3.h,
                                                       (short)0, acc3, false, false);

        if (more) {
            STAGE_WAIT();            // async LDS writes landed
        }
        __syncthreads();             // all waves done reading cur before it is reused
        a_cur = a_next;
    }

    // C/D layout: element i of v8f, lane L -> row m0 + (L>>4)*8 + i, col n0 + t*16 + (L&15)
#pragma unroll
    for (int t = 0; t < 4; ++t) {
        v8f c = (t == 0) ? acc0 : (t == 1) ? acc1 : (t == 2) ? acc2 : acc3;
#pragma unroll
        for (int i = 0; i < 8; ++i) {
            int m = m0 + lhalf * 8 + i;
            int n = n0 + t * 16 + l16;
            act[m * KD + n] = c[i];
        }
    }
}

// Kernel 4: feats[b,k] = sum_{b'} exp(-sum_d |act[b,k,d]-act[b',k,d]|)
// grid = (K=64, 8). Block stages act[:, k, :] (512x16 f32 = 32KB) in LDS.
// 4 consecutive lanes share one b; each covers 128 b' values; quad shfl reduction.
__global__ __launch_bounds__(256) void feats_kernel(const float* __restrict__ act,
                                                    float* __restrict__ out) {
    __shared__ float sact[Bsz * Dsz];   // 32 KB
    const int k = blockIdx.x;

    for (int idx = threadIdx.x; idx < Bsz * 4; idx += 256) {
        int b = idx >> 2;
        int q = idx & 3;
        float4 v = *(const float4*)(act + b * KD + k * Dsz + q * 4);
        *(float4*)(sact + b * Dsz + q * 4) = v;
    }
    __syncthreads();

    const int b       = blockIdx.y * 64 + (threadIdx.x >> 2);
    const int quarter = threadIdx.x & 3;

    const float* mine = sact + b * Dsz;
    float4 m0 = *(const float4*)(mine + 0);
    float4 m1 = *(const float4*)(mine + 4);
    float4 m2 = *(const float4*)(mine + 8);
    float4 m3 = *(const float4*)(mine + 12);

    float sum = 0.f;
    for (int bp = quarter * 128; bp < quarter * 128 + 128; ++bp) {
        const float* o = sact + bp * Dsz;
        float4 o0 = *(const float4*)(o + 0);
        float4 o1 = *(const float4*)(o + 4);
        float4 o2 = *(const float4*)(o + 8);
        float4 o3 = *(const float4*)(o + 12);
        float s = 0.f;
        s += fabsf(m0.x - o0.x); s += fabsf(m0.y - o0.y);
        s += fabsf(m0.z - o0.z); s += fabsf(m0.w - o0.w);
        s += fabsf(m1.x - o1.x); s += fabsf(m1.y - o1.y);
        s += fabsf(m1.z - o1.z); s += fabsf(m1.w - o1.w);
        s += fabsf(m2.x - o2.x); s += fabsf(m2.y - o2.y);
        s += fabsf(m2.z - o2.z); s += fabsf(m2.w - o2.w);
        s += fabsf(m3.x - o3.x); s += fabsf(m3.y - o3.y);
        s += fabsf(m3.z - o3.z); s += fabsf(m3.w - o3.w);
        sum += __expf(-s);
    }
    sum += __shfl_xor(sum, 1);
    sum += __shfl_xor(sum, 2);
    if (quarter == 0) {
        out[b * OUTC + Fsz + k] = sum;
    }
}

extern "C" void kernel_launch(void* const* d_in, const int* in_sizes, int n_in,
                              void* d_out, int out_size, void* d_ws, size_t ws_size,
                              hipStream_t stream) {
    const float* x = (const float*)d_in[0];        // [512, 2048] f32
    const float* W = (const float*)d_in[1];        // [2048, 64, 16] f32
    float* out = (float*)d_out;                    // [512, 2112] f32

    // Workspace layout: xb bf16 (2MB) | wT bf16 (4MB) | act f32 (2MB)
    unsigned short* xb  = (unsigned short*)d_ws;
    unsigned short* wT  = (unsigned short*)((char*)d_ws + (2u << 20));
    float*          act = (float*)((char*)d_ws + (6u << 20));

    // 1) convert x -> bf16, copy x -> out[:, :2048]
    prep_kernel<<<(Bsz * Fsz / 4) / 256, 256, 0, stream>>>(x, xb, out);

    // 2) W f32 [2048,1024] -> wT bf16 [1024,2048]
    {
        dim3 grid(KD / 32, Fsz / 32);
        dim3 block(32, 8);
        convert_transpose_kernel<<<grid, block, 0, stream>>>(W, wT);
    }

    // 3) GEMM via WMMA bf16: act = x @ W
    {
        dim3 grid(Bsz / 128, KD / 64);
        gemm_wmma_kernel<<<grid, 256, 0, stream>>>(xb, wT, act);
    }

    // 4) pairwise L1 + exp-sum -> out[:, 2048:2112]
    {
        dim3 grid(Ksz, Bsz / 64);
        feats_kernel<<<grid, 256, 0, stream>>>(act, out);
    }
}
